// MyLoss_43293270344002
// MI455X (gfx1250) — matrix operand order repaired and checked by reading, not compiled, not run on hardware
//
#include <hip/hip_runtime.h>
#include <hip/hip_fp16.h>
#include <hip/hip_bf16.h>

// ---------------------------------------------------------------------------
// ArcFace loss for MI455X (gfx1250, wave32, WMMA).
//   input  [512,512] f32, weight [100000,512] f32, label [512] int
//   out    = concat( output[512,100000] f32 , weight_loss[1] f32 )
// Strategy: one fused [1024 x 100000] f16 GEMM (A = [xn ; wn[label]]) against
// wn^T using v_wmma_f32_16x16x32_f16; wn (f16, 102.4MB) stays resident in the
// 192MB L2 so HBM traffic ~0.5GB total -> memory bound (~22us @ 23.3TB/s).
// ---------------------------------------------------------------------------

typedef __attribute__((ext_vector_type(16))) _Float16 v16h;
typedef __attribute__((ext_vector_type(8)))  _Float16 v8h;
typedef __attribute__((ext_vector_type(8)))  float    v8f;

#define IN_F   512
#define N_OUT  100000
#define BATCH  512
#define AROWS  1024
#define NXB    ((N_OUT + 127) / 128)   // 782 N-blocks of 128 cols

#define S_SCALE 30.0f
#define COS_M_C 0.95533648912560602f   // cos(0.3)
#define SIN_M_C 0.29552020666133955f   // sin(0.3)

// ---------------------------------------------------------------------------
// Row L2-normalize (512 floats/row) -> f16. One block per row, 256 threads.
// ---------------------------------------------------------------------------
__device__ __forceinline__ void normalize_row_f16(const float* __restrict__ src,
                                                  _Float16* __restrict__ dst) {
    const int t = threadIdx.x;
    const float x0 = src[t];
    const float x1 = src[t + 256];
    float ss = x0 * x0 + x1 * x1;
    #pragma unroll
    for (int off = 16; off > 0; off >>= 1) ss += __shfl_down(ss, off, 32);
    __shared__ float lds[8];
    if ((t & 31) == 0) lds[t >> 5] = ss;
    __syncthreads();
    float n = 0.0f;
    #pragma unroll
    for (int w = 0; w < 8; ++w) n += lds[w];
    const float r = 1.0f / fmaxf(sqrtf(n), 1e-12f);
    dst[t]       = (_Float16)(x0 * r);
    dst[t + 256] = (_Float16)(x1 * r);
}

// A rows 0..511 = normalize(input); rows 512..1023 = normalize(weight[label])
__global__ __launch_bounds__(256)
void build_a_kernel(const float* __restrict__ inp, const float* __restrict__ W,
                    const int* __restrict__ label, _Float16* __restrict__ A) {
    const int row = blockIdx.x;                        // 0..1023
    const float* src = (row < BATCH)
        ? inp + (size_t)row * IN_F
        : W + (size_t)label[row - BATCH] * IN_F;
    normalize_row_f16(src, A + (size_t)row * IN_F);
}

__global__ __launch_bounds__(256)
void norm_w_kernel(const float* __restrict__ W, _Float16* __restrict__ Wn) {
    const int row = blockIdx.x;                        // 0..99999
    normalize_row_f16(W + (size_t)row * IN_F, Wn + (size_t)row * IN_F);
}

// ---------------------------------------------------------------------------
// Fused GEMM + epilogue.
// Grid: x = NXB (128 cols each), y = 16 (64 A-rows each). Block = 256 (8 waves).
// Each wave: 64M x 16N strip = 4 x (16x16) accumulators, K=512 in steps of 32.
// ISA fragment layouts (CDNA5 ISA):
//   A 16x32 f16 : lane -> M = lane%16, halves = K in {kb..kb+7, kb+16..kb+23},
//                 kb = 8*(lane/16)
//   B 32x16 f16 : lane -> N = lane%16, ISA K order as A; here A and B both load K = 16*(lane/16)+0..15
//   C 16x16 f32 : vgpr r, lane -> row = r + 8*(lane/16), col = lane%16
// ---------------------------------------------------------------------------
__global__ __launch_bounds__(256)
void arcface_gemm_kernel(const _Float16* __restrict__ A,
                         const _Float16* __restrict__ Wn,
                         const int*      __restrict__ label,
                         float*          __restrict__ out,
                         float*          __restrict__ minpartial) {
    const int lane  = threadIdx.x & 31;
    const int wave  = threadIdx.x >> 5;
    const int half  = lane >> 4;
    const int l16   = lane & 15;
    const int mbase = blockIdx.y * 64;
    const int col   = blockIdx.x * 128 + wave * 16 + l16;
    const bool colOK = (col < N_OUT);
    const int colc  = colOK ? col : (N_OUT - 1);       // clamp: keep EXEC all-1s

    // B fragment base: row colc of Wn, K offset 16*half (contiguous 16 halves)
    const _Float16* bp = Wn + (size_t)colc * IN_F + half * 16;
    // A fragment bases per M-subtile
    const _Float16* ap[4];
    #pragma unroll
    for (int mt = 0; mt < 4; ++mt)
        ap[mt] = A + (size_t)(mbase + mt * 16 + l16) * IN_F + half * 16;

    v8f acc[4] = {v8f{}, v8f{}, v8f{}, v8f{}};

    #pragma unroll
    for (int kk = 0; kk < IN_F; kk += 32) {
        __builtin_prefetch(bp + kk + 256, 0, 1);       // global_prefetch_b8
        const v16h b = *(const v16h*)(bp + kk);
        #pragma unroll
        for (int mt = 0; mt < 4; ++mt) {
            const v8h alo = *(const v8h*)(ap[mt] + kk);
            const v8h ahi = *(const v8h*)(ap[mt] + kk + 8);
            const v16h a = __builtin_shufflevector(alo, ahi,
                0, 1, 2, 3, 4, 5, 6, 7, 8, 9, 10, 11, 12, 13, 14, 15);
            acc[mt] = __builtin_amdgcn_wmma_f32_16x16x32_f16(
                /*neg_a=*/false, a, /*neg_b=*/false, b,
                /*c_mod=*/(short)0, acc[mt],
                /*reuse_a=*/false, /*reuse_b=*/false);
        }
    }

    __shared__ float smin[8][64];

    if (blockIdx.y < 8) {
        // ---- cosine rows: margin epilogue + output store ----
        #pragma unroll
        for (int mt = 0; mt < 4; ++mt) {
            #pragma unroll
            for (int r = 0; r < 8; ++r) {
                const int row = mbase + mt * 16 + half * 8 + r;
                const float c  = acc[mt][r];
                float s2 = 1.0f - c * c;
                s2 = fminf(fmaxf(s2, 0.0f), 1.0f);
                const float phi0 = c * COS_M_C - sqrtf(s2) * SIN_M_C;
                const float phi  = (c > 0.0f) ? phi0 : c;   // easy margin
                const float v    = (col == label[row]) ? phi : c;
                if (colOK) out[(size_t)row * N_OUT + col] = v * S_SCALE;
            }
        }
    } else {
        // ---- d2 rows: per-row min over this block's 128 cols ----
        #pragma unroll
        for (int mt = 0; mt < 4; ++mt) {
            #pragma unroll
            for (int r = 0; r < 8; ++r) {
                const int row = mbase + mt * 16 + half * 8 + r;  // 512..1023
                const int lab = label[row - BATCH];
                const float d2 = 2.0f - 2.0f * acc[mt][r];
                float v = (colOK && col != lab) ? d2 : 1000.0f;  // one-hot mask
                #pragma unroll
                for (int m = 1; m < 16; m <<= 1)
                    v = fminf(v, __shfl_xor(v, m, 32));          // min over 16 cols
                if (l16 == 0) smin[wave][mt * 16 + half * 8 + r] = v;
            }
        }
        __syncthreads();
        const int t = threadIdx.x;
        if (t < 64) {
            float v = smin[0][t];
            #pragma unroll
            for (int w = 1; w < 8; ++w) v = fminf(v, smin[w][t]);
            minpartial[(size_t)(mbase - BATCH + t) * NXB + blockIdx.x] = v;
        }
    }
}

// ---------------------------------------------------------------------------
// weight_loss = BATCH / sum_row( min_b minpartial[row][b] )
// ---------------------------------------------------------------------------
__global__ __launch_bounds__(256)
void finalize_kernel(const float* __restrict__ minpartial, float* __restrict__ outv) {
    const int t = threadIdx.x;
    float sum = 0.0f;
    for (int row = t; row < BATCH; row += 256) {
        float m = 3.402823466e38f;
        for (int b = 0; b < NXB; ++b)
            m = fminf(m, minpartial[(size_t)row * NXB + b]);
        sum += m;
    }
    #pragma unroll
    for (int off = 16; off > 0; off >>= 1) sum += __shfl_down(sum, off, 32);
    __shared__ float lds[8];
    if ((t & 31) == 0) lds[t >> 5] = sum;
    __syncthreads();
    if (t == 0) {
        float tot = 0.0f;
        #pragma unroll
        for (int w = 0; w < 8; ++w) tot += lds[w];
        outv[0] = (float)BATCH / tot;
    }
}

// ---------------------------------------------------------------------------
extern "C" void kernel_launch(void* const* d_in, const int* in_sizes, int n_in,
                              void* d_out, int out_size, void* d_ws, size_t ws_size,
                              hipStream_t stream) {
    const float* inp = (const float*)d_in[0];
    const float* W   = (const float*)d_in[1];
    const int*   lab = (const int*)d_in[2];
    float* out = (float*)d_out;

    // Workspace layout (all 256B aligned):
    //   A   f16 [1024 x 512]            1,048,576 B
    //   Wn  f16 [100000 x 512]        102,400,000 B
    //   minpartial f32 [512 x NXB]      1,601,536 B
    char* ws = (char*)d_ws;
    _Float16* A    = (_Float16*)ws;
    _Float16* Wn   = (_Float16*)(ws + (size_t)(1 << 20));
    float*    minp = (float*)(ws + (size_t)(1 << 20) + 102400000u);

    build_a_kernel<<<AROWS, 256, 0, stream>>>(inp, W, lab, A);
    norm_w_kernel<<<N_OUT, 256, 0, stream>>>(W, Wn);

    dim3 grid(NXB, 16);
    arcface_gemm_kernel<<<grid, 256, 0, stream>>>(A, Wn, lab, out, minp);

    finalize_kernel<<<1, 256, 0, stream>>>(minp, out + (size_t)BATCH * N_OUT);
}
